// STU_41523743817874
// MI455X (gfx1250) — compile-verified, matrix operand order checked by reading, not run on hardware
//
#include <hip/hip_runtime.h>

// ---- problem constants (from reference) ------------------------------------
#define B_   2
#define L_   1024
#define D_   768
#define K_   16
#define SK_  32            // 2 signs * 16 filters
#define BLT  (B_ * L_)     // 2048 total rows
#define NDELTA 64          // Toeplitz tile diagonal offsets (delta/16)
#define LDSTRIDE 776       // 768 + 8 pad: kills LDS bank conflicts on row reads

typedef __attribute__((ext_vector_type(16))) unsigned short v16u;
typedef __attribute__((ext_vector_type(8)))  unsigned short v8u;
typedef __attribute__((ext_vector_type(16))) __bf16         v16bf;
typedef __attribute__((ext_vector_type(8)))  float          v8f;

__device__ __forceinline__ unsigned short f2bf(float f) {
  unsigned int u = __builtin_bit_cast(unsigned int, f);
  u += 0x7FFFu + ((u >> 16) & 1u);              // round-to-nearest-even
  return (unsigned short)(u >> 16);
}

__device__ __forceinline__ v8f zero8() {
  v8f z = {0.f, 0.f, 0.f, 0.f, 0.f, 0.f, 0.f, 0.f};
  return z;
}

__device__ __forceinline__ v8f wmma_bf16(v16u a, v16u b, v8f c) {
  return __builtin_amdgcn_wmma_f32_16x16x32_bf16(
      /*neg_a=*/false, __builtin_bit_cast(v16bf, a),
      /*neg_b=*/false, __builtin_bit_cast(v16bf, b),
      /*c_mod=*/(short)0, c, /*reuse_a=*/false, /*reuse_b=*/false);
}

// Two contiguous 16-byte runs -> one 16-element bf16 fragment (lands in
// consecutive VGPRs; lowers to two b128 loads, global or ds).
__device__ __forceinline__ v16u ldpair(const unsigned short* p0,
                                       const unsigned short* p1) {
  union { v16u v; v8u h[2]; } u;
  u.h[0] = *(const v8u*)p0;
  u.h[1] = *(const v8u*)p1;
  return u.v;
}

// Slot j (0..15) of a lane's 16-bf16 WMMA A/B fragment -> K index within the
// 32-wide contraction block (CDNA5 16-bit operand layout, ISA 7.12.2):
// lanes 0-15 hold K {0..7, 16..23}; lanes 16-31 hold K {8..15, 24..31}.
__device__ __forceinline__ int kslot(int lane, int j) {
  int hi = (lane >> 4) & 1;
  return (j < 8) ? (j + hi * 8) : (j + 8 + hi * 8);
}

// ---- prep 0: Toeplitz phi tiles in WMMA-A lane layout ----------------------
// At[((sk*64 + delta/16)*32 + lane)*16 + j] = phi_sign[delta + M(lane) - K(lane,j)]
__global__ void prep_atile(const float* __restrict__ phi,
                           unsigned short* __restrict__ At) {
  int tid  = blockIdx.x * 256 + threadIdx.x;   // SK_*NDELTA*32*16 = 1,048,576
  int j    = tid & 15;
  int lane = (tid >> 4) & 31;
  int dix  = (tid >> 9) & 63;
  int sk   = tid >> 15;
  int k    = sk & 15;
  int sign = sk >> 4;
  int l    = lane & 15;
  int idx  = dix * 16 + l - kslot(lane, j);
  float v = 0.f;
  if (idx >= 0 && idx < L_) {
    v = phi[idx * K_ + k];
    if (sign && (idx & 1)) v = -v;               // phi_minus[t] = (-1)^t phi[t]
  }
  At[tid] = f2bf(v);
}

// ---- prep 1: x -> bf16, transposed to [b][d][l] ----------------------------
__global__ void prep_ut(const float* __restrict__ x,
                        unsigned short* __restrict__ uT) {
  int tid = blockIdx.x * 256 + threadIdx.x;     // (b*D + d)*L + l
  int l  = tid & (L_ - 1);
  int bd = tid >> 10;
  int d  = bd % D_;
  int b  = bd / D_;
  uT[tid] = f2bf(x[((size_t)(b * L_ + l)) * D_ + d]);
}

// ---- prep 2: M_plus|M_minus -> bf16 [sk][o][i] -----------------------------
__global__ void prep_m(const float* __restrict__ Mp, const float* __restrict__ Mm,
                       unsigned short* __restrict__ Mb) {
  size_t tid = (size_t)blockIdx.x * 256 + threadIdx.x;  // SK_*D_*D_
  int i  = (int)(tid % D_);
  size_t t2 = tid / D_;
  int o  = (int)(t2 % D_);
  int sk = (int)(t2 / D_);
  int k = sk & 15, sign = sk >> 4;
  const float* src = sign ? Mm : Mp;
  Mb[tid] = f2bf(src[((size_t)k * D_ + o) * D_ + i]);
}

// ---- fused main: causal conv (Toeplitz WMMA) -> LDS -> contraction WMMA ----
__global__ __launch_bounds__(256, 1) void stu_main(
    const unsigned short* __restrict__ At,
    const unsigned short* __restrict__ uT,
    const unsigned short* __restrict__ Mb,
    float* __restrict__ part) {
  __shared__ __align__(16) unsigned short lds[32 * LDSTRIDE];  // 48.5 KB U-tile
  const int lane  = threadIdx.x & 31;
  const int w     = threadIdx.x >> 5;        // wave 0..7
  const int ln    = lane & 15;
  const int half  = lane >> 4;
  const int base0 = half * 8;                // run start per 16-bit operand layout
  const int BL0   = blockIdx.x * 32;         // 32 bl-rows per workgroup
  const int b     = BL0 / L_;                // never straddles batch
  const int Lb0   = BL0 % L_;                // multiple of 32
  const int g     = blockIdx.y;              // (sign,k) group 0..3

  // Lb0 % 32 == 0  =>  both 16-row l-tiles share the same causal s-block
  // count, and their Toeplitz diagonals are dix0 and dix0+1.
  const int ns = (Lb0 >> 5) + 1;

  v8f acc[2][6];                             // 2 l-tiles x 6 o-tiles per wave
#pragma unroll
  for (int lb = 0; lb < 2; ++lb)
#pragma unroll
    for (int j = 0; j < 6; ++j) acc[lb][j] = zero8();

  for (int p = 0; p < 8; ++p) {
    const int sk = g * 8 + p;
    if (p) __syncthreads();                  // LDS reuse across iterations

    // phase 1: U_tile[32 x 768] = Toeplitz(phi_sk) * x.  j outermost keeps only
    // ua0/ua1 live; x-fragment B shared across both l-tiles (2 WMMAs / B load).
#pragma unroll 1
    for (int j = 0; j < 6; ++j) {
      const int d0 = (w * 6 + j) * 16 + ln;
      const unsigned short* uTd = uT + ((size_t)(b * D_ + d0) << 10) + base0;
      v8f ua0 = zero8(), ua1 = zero8();
#pragma unroll 1
      for (int si = 0; si < ns; ++si) {
        const int S0   = si * 32;
        const int dix0 = (Lb0 - S0) >> 4;
        const unsigned short* ap = At + ((((sk * NDELTA) + dix0) * 32 + lane) << 4);
        const v16u a0 = *(const v16u*)ap;          // diagonal for l-tile 0
        const v16u a1 = *(const v16u*)(ap + 512);  // dix0+1 -> l-tile 1
        const v16u bb = ldpair(uTd + S0, uTd + S0 + 16);
        ua0 = wmma_bf16(a0, bb, ua0);
        ua1 = wmma_bf16(a1, bb, ua1);
      }
      // C layout: VGPR r -> row r + 8*half
#pragma unroll
      for (int r = 0; r < 8; ++r) {
        lds[(r + base0) * LDSTRIDE + d0]      = f2bf(ua0[r]);
        lds[(16 + r + base0) * LDSTRIDE + d0] = f2bf(ua1[r]);
      }
    }
    __syncthreads();

    // phase 2: out[bl,o] += U_tile[bl,i] * M[sk][o][i].  Both LDS A-fragments
    // hoisted per ib; each M-fragment feeds 2 WMMAs.
#pragma unroll 1
    for (int ib = 0; ib < 24; ++ib) {
      const unsigned short* lp0 = &lds[ln * LDSTRIDE + ib * 32 + base0];
      const unsigned short* lp1 = lp0 + 16 * LDSTRIDE;
      const v16u a0 = ldpair(lp0, lp0 + 16);
      const v16u a1 = ldpair(lp1, lp1 + 16);
#pragma unroll
      for (int j = 0; j < 6; ++j) {
        const int o0 = (w * 6 + j) * 16 + ln;
        const unsigned short* mp = Mb + (size_t)(sk * D_ + o0) * D_ + ib * 32 + base0;
        __builtin_prefetch(mp + 32, 0, 1);
        const v16u bb = ldpair(mp, mp + 16);
        acc[0][j] = wmma_bf16(a0, bb, acc[0][j]);
        acc[1][j] = wmma_bf16(a1, bb, acc[1][j]);
      }
    }
  }

  // per-group partial (summed by reduce_out; avoids fp atomics -> deterministic)
  float* pg = part + (size_t)g * BLT * D_;
#pragma unroll
  for (int lb = 0; lb < 2; ++lb)
#pragma unroll
    for (int j = 0; j < 6; ++j) {
      const int col = (w * 6 + j) * 16 + ln;
#pragma unroll
      for (int r = 0; r < 8; ++r) {
        const int row = BL0 + lb * 16 + r + base0;
        pg[(size_t)row * D_ + col] = acc[lb][j][r];
      }
    }
}

__global__ void reduce_out(const float* __restrict__ part, float* __restrict__ out) {
  int tid = blockIdx.x * 256 + threadIdx.x;
  const size_t N = (size_t)BLT * D_;
  out[tid] = part[tid] + part[N + tid] + part[2 * N + tid] + part[3 * N + tid];
}

extern "C" void kernel_launch(void* const* d_in, const int* in_sizes, int n_in,
                              void* d_out, int out_size, void* d_ws, size_t ws_size,
                              hipStream_t stream) {
  (void)in_sizes; (void)n_in; (void)out_size; (void)ws_size;
  const float* x   = (const float*)d_in[0];
  const float* phi = (const float*)d_in[1];
  const float* Mp  = (const float*)d_in[2];
  const float* Mm  = (const float*)d_in[3];
  // d_in[4] = n (FFT length) unused: conv done exactly in time domain.

  char* ws = (char*)d_ws;
  size_t off = 0;
  unsigned short* At = (unsigned short*)(ws + off); off += (size_t)SK_ * NDELTA * 32 * 16 * 2; // 2 MiB
  unsigned short* uT = (unsigned short*)(ws + off); off += (size_t)B_ * D_ * L_ * 2;           // 3 MiB
  unsigned short* Mb = (unsigned short*)(ws + off); off += (size_t)SK_ * D_ * D_ * 2;          // 36 MiB
  float*          pt = (float*)(ws + off);          off += (size_t)4 * BLT * D_ * 4;           // 24 MiB

  prep_atile<<<(SK_ * NDELTA * 32 * 16) / 256, 256, 0, stream>>>(phi, At);
  prep_ut   <<<(B_ * D_ * L_) / 256,            256, 0, stream>>>(x, uT);
  prep_m    <<<(SK_ * D_ * D_) / 256,           256, 0, stream>>>(Mp, Mm, Mb);
  stu_main  <<<dim3(BLT / 32, 4),               256, 0, stream>>>(At, uT, Mb, pt);
  reduce_out<<<(BLT * D_) / 256,                256, 0, stream>>>(pt, (float*)d_out);
}
